// TwoLayerHeteroSAGE_65721589563817
// MI455X (gfx1250) — matrix operand order repaired and checked by reading, not compiled, not run on hardware
//
#include <hip/hip_runtime.h>
#include <math.h>

typedef __attribute__((ext_vector_type(16))) __bf16 v16bf;
typedef __attribute__((ext_vector_type(8)))  float  v8f;
typedef unsigned short u16t;

#define BM 64      // M-tile (smaller tile -> 128 blocks, fills more WGPs)
#define BN 256     // full N per block: each adjacency row read once from HBM
#define BK 32
#define NT 128     // 4 waves per block, wave tile 64x64 (1 x 4 arrangement)

__device__ __forceinline__ u16t f2bf(float f) {
    unsigned u = __float_as_uint(f);
    unsigned rb = ((u >> 16) & 1u) + 0x7FFFu;   // round-to-nearest-even
    return (u16t)((u + rb) >> 16);
}
__device__ __forceinline__ float bf2f(u16t u) {
    return __uint_as_float(((unsigned)u) << 16);
}
// rounding pack (real-valued data)
__device__ __forceinline__ unsigned pk2(float lo, float hi) {
    return (unsigned)f2bf(lo) | ((unsigned)f2bf(hi) << 16);
}
// truncating pack: exact for binary {0,1} adjacency, folds to v_perm_b32
__device__ __forceinline__ unsigned pk2t(float lo, float hi) {
    return (__float_as_uint(hi) & 0xFFFF0000u) | (__float_as_uint(lo) >> 16);
}

union FragU { v16bf v; uint4 q[2]; };

// ---------------------------------------------------------------------------
// f32 -> bf16 cast (grid-stride)
// ---------------------------------------------------------------------------
__global__ void cast_bf16_kernel(const float* __restrict__ src,
                                 u16t* __restrict__ dst, size_t n) {
    size_t i = (size_t)blockIdx.x * blockDim.x + threadIdx.x;
    size_t stride = (size_t)gridDim.x * blockDim.x;
    for (; i < n; i += stride) dst[i] = f2bf(src[i]);
}

// ---------------------------------------------------------------------------
// Y = X(MxK bf16) @ W(NxK bf16)^T + bias(N f32)
// colmajor == 0 : Y row-major (M x N)
// colmajor == 1 : Y N-major (N x M), K-contiguous for the next GEMM;
//                 8 rows per lane contiguous -> one b128 store per tile.
// WG tile 64x256, 4 waves, wave tile 64x64, K-step 32.
// ---------------------------------------------------------------------------
__global__ __launch_bounds__(NT) void lin_kernel(
    const u16t* __restrict__ X, const u16t* __restrict__ W,
    const float* __restrict__ bias, u16t* __restrict__ Y,
    int M, int N, int K, int colmajor)
{
    __shared__ u16t As[BM][BK];    // 4 KB, row-major (M x K)
    __shared__ u16t Bts[BN][BK];   // 16 KB, (N x K)

    const int t    = threadIdx.x;
    const int lane = t & 31;
    const int wave = t >> 5;
    const int wn   = wave * 64;
    const int m0   = blockIdx.x * BM;

    v8f acc[4][4];
    const v8f vz = {0.f,0.f,0.f,0.f,0.f,0.f,0.f,0.f};
#pragma unroll
    for (int i = 0; i < 4; i++)
#pragma unroll
        for (int j = 0; j < 4; j++) acc[i][j] = vz;

    for (int k0 = 0; k0 < K; k0 += BK) {
        // A tile: thread -> (row t>>1, half (t&1)*16), 2 x b128
        {
            int r = t >> 1, h = (t & 1) * 16;
            const uint4* s = (const uint4*)(X + (size_t)(m0 + r) * K + k0 + h);
            uint4 x0 = s[0], x1 = s[1];
            *(uint4*)&As[r][h]     = x0;
            *(uint4*)&As[r][h + 8] = x1;
        }
        // B tile: W rows t and t+128 are K-contiguous, straight b128 copies
#pragma unroll
        for (int nn = 0; nn < BN; nn += NT) {
            const uint4* s = (const uint4*)(W + (size_t)(nn + t) * K + k0);
            uint4 b0 = s[0], b1 = s[1], b2 = s[2], b3 = s[3];
            uint4* d = (uint4*)&Bts[nn + t][0];
            d[0] = b0; d[1] = b1; d[2] = b2; d[3] = b3;
        }
        __syncthreads();

        FragU a[4], b[4];
        const int arow = lane & 15;
        const int kb   = (lane >> 4) * 8;
        const int rowb = (lane >> 4) * 16;
#pragma unroll
        for (int i = 0; i < 4; i++) {
            int r = i * 16 + arow;
            a[i].q[0] = *(const uint4*)&As[r][kb];
            a[i].q[1] = *(const uint4*)&As[r][16 + kb];
        }
#pragma unroll
        for (int j = 0; j < 4; j++) {
            int c = wn + j * 16 + (lane & 15);
            b[j].q[0] = *(const uint4*)&Bts[c][kb];
            b[j].q[1] = *(const uint4*)&Bts[c][16 + kb];
        }
#pragma unroll
        for (int i = 0; i < 4; i++)
#pragma unroll
            for (int j = 0; j < 4; j++)
                acc[i][j] = __builtin_amdgcn_wmma_f32_16x16x32_bf16(
                    false, a[i].v, false, b[j].v, (short)0, acc[i][j],
                    false, false);
        __syncthreads();
    }

    const int cn = lane & 15;
    const int rb = (lane >> 4) * 8;
    if (colmajor) {
#pragma unroll
        for (int i = 0; i < 4; i++)
#pragma unroll
            for (int j = 0; j < 4; j++) {
                int col = wn + j * 16 + cn;
                float bv = bias[col];
                uint4 p;
                p.x = pk2(acc[i][j][0] + bv, acc[i][j][1] + bv);
                p.y = pk2(acc[i][j][2] + bv, acc[i][j][3] + bv);
                p.z = pk2(acc[i][j][4] + bv, acc[i][j][5] + bv);
                p.w = pk2(acc[i][j][6] + bv, acc[i][j][7] + bv);
                int row0 = m0 + i * 16 + rb;
                *(uint4*)(Y + (size_t)col * M + row0) = p;
            }
    } else {
#pragma unroll
        for (int i = 0; i < 4; i++)
#pragma unroll
            for (int j = 0; j < 4; j++) {
                int col = wn + j * 16 + cn;
                float bv = bias[col];
#pragma unroll
                for (int r = 0; r < 8; r++) {
                    int row = m0 + i * 16 + rb + r;
                    Y[(size_t)row * N + col] = f2bf(acc[i][j][r] + bv);
                }
            }
    }
}

// ---------------------------------------------------------------------------
// D = leaky_relu( Self + A1 @ B1 + A2 @ B2 ),   slope 0.1
// A1,A2: MxK f32 (binary adjacency; truncating-pack to bf16 during staging)
// B1,B2: N-major bf16 (N x K, K-contiguous) -- produced by lin_kernel colmajor
// Self : MxN bf16 (biases already folded into B/Self).
// Output: Yb (bf16 row-major, may be null) and/or Yf (f32 row-major, may be null).
// ---------------------------------------------------------------------------
__global__ __launch_bounds__(NT) void agg_kernel(
    const float* __restrict__ A1, const float* __restrict__ A2,
    const u16t* __restrict__ B1, const u16t* __restrict__ B2,
    const u16t* __restrict__ Self,
    u16t* __restrict__ Yb, float* __restrict__ Yf,
    int M, int N, int K)
{
    __shared__ u16t A1s[BM][BK];   // 4 KB
    __shared__ u16t A2s[BM][BK];   // 4 KB
    __shared__ u16t B1ts[BN][BK];  // 16 KB
    __shared__ u16t B2ts[BN][BK];  // 16 KB

    const int t    = threadIdx.x;
    const int lane = t & 31;
    const int wave = t >> 5;
    const int wn   = wave * 64;
    const int m0   = blockIdx.x * BM;

    v8f acc[4][4];
    const v8f vz = {0.f,0.f,0.f,0.f,0.f,0.f,0.f,0.f};
#pragma unroll
    for (int i = 0; i < 4; i++)
#pragma unroll
        for (int j = 0; j < 4; j++) acc[i][j] = vz;

    const int r = t >> 1, h = (t & 1) * 16;          // A staging coords
    const float* a1row = A1 + (size_t)(m0 + r) * K + h;
    const float* a2row = A2 + (size_t)(m0 + r) * K + h;

    for (int k0 = 0; k0 < K; k0 += BK) {
        // prefetch adjacency two K-steps ahead (the HBM-critical stream)
        if (k0 + 2 * BK < K) {
            __builtin_prefetch(a1row + k0 + 2 * BK, 0, 3);
            __builtin_prefetch(a2row + k0 + 2 * BK, 0, 3);
        }
        // A tiles: 16 f32 per thread per matrix, truncate+pack -> 2 b128 stores
        {
            const float4* s1 = (const float4*)(a1row + k0);
            const float4* s2 = (const float4*)(a2row + k0);
            float4 f0 = s1[0], f1 = s1[1], f2 = s1[2], f3 = s1[3];
            uint4 p;
            p.x = pk2t(f0.x, f0.y); p.y = pk2t(f0.z, f0.w);
            p.z = pk2t(f1.x, f1.y); p.w = pk2t(f1.z, f1.w);
            *(uint4*)&A1s[r][h] = p;
            p.x = pk2t(f2.x, f2.y); p.y = pk2t(f2.z, f2.w);
            p.z = pk2t(f3.x, f3.y); p.w = pk2t(f3.z, f3.w);
            *(uint4*)&A1s[r][h + 8] = p;
            f0 = s2[0]; f1 = s2[1]; f2 = s2[2]; f3 = s2[3];
            p.x = pk2t(f0.x, f0.y); p.y = pk2t(f0.z, f0.w);
            p.z = pk2t(f1.x, f1.y); p.w = pk2t(f1.z, f1.w);
            *(uint4*)&A2s[r][h] = p;
            p.x = pk2t(f2.x, f2.y); p.y = pk2t(f2.z, f2.w);
            p.z = pk2t(f3.x, f3.y); p.w = pk2t(f3.z, f3.w);
            *(uint4*)&A2s[r][h + 8] = p;
        }
        // B tiles: N-major layout, rows t and t+128 K-contiguous -> b128 copies
#pragma unroll
        for (int nn = 0; nn < BN; nn += NT) {
            const uint4* s = (const uint4*)(B1 + (size_t)(nn + t) * K + k0);
            uint4 b0 = s[0], b1 = s[1], b2 = s[2], b3 = s[3];
            uint4* d = (uint4*)&B1ts[nn + t][0];
            d[0] = b0; d[1] = b1; d[2] = b2; d[3] = b3;
            s = (const uint4*)(B2 + (size_t)(nn + t) * K + k0);
            b0 = s[0]; b1 = s[1]; b2 = s[2]; b3 = s[3];
            d = (uint4*)&B2ts[nn + t][0];
            d[0] = b0; d[1] = b1; d[2] = b2; d[3] = b3;
        }
        __syncthreads();

        FragU a1[4], a2[4], b;
        const int arow = lane & 15;
        const int kb   = (lane >> 4) * 8;
        const int rowb = (lane >> 4) * 16;
#pragma unroll
        for (int i = 0; i < 4; i++) {
            int rr = i * 16 + arow;
            a1[i].q[0] = *(const uint4*)&A1s[rr][kb];
            a1[i].q[1] = *(const uint4*)&A1s[rr][16 + kb];
            a2[i].q[0] = *(const uint4*)&A2s[rr][kb];
            a2[i].q[1] = *(const uint4*)&A2s[rr][16 + kb];
        }
#pragma unroll
        for (int j = 0; j < 4; j++) {
            int c = wn + j * 16 + (lane & 15);
            b.q[0] = *(const uint4*)&B1ts[c][kb];
            b.q[1] = *(const uint4*)&B1ts[c][16 + kb];
#pragma unroll
            for (int i = 0; i < 4; i++)
                acc[i][j] = __builtin_amdgcn_wmma_f32_16x16x32_bf16(
                    false, a1[i].v, false, b.v, (short)0, acc[i][j],
                    false, false);
            b.q[0] = *(const uint4*)&B2ts[c][kb];
            b.q[1] = *(const uint4*)&B2ts[c][16 + kb];
#pragma unroll
            for (int i = 0; i < 4; i++)
                acc[i][j] = __builtin_amdgcn_wmma_f32_16x16x32_bf16(
                    false, a2[i].v, false, b.v, (short)0, acc[i][j],
                    false, false);
        }
        __syncthreads();
    }

    const int cn = lane & 15;
    const int rb = (lane >> 4) * 8;
#pragma unroll
    for (int i = 0; i < 4; i++)
#pragma unroll
        for (int j = 0; j < 4; j++) {
            int col = wn + j * 16 + cn;
#pragma unroll
            for (int rr = 0; rr < 8; rr++) {
                int row = m0 + i * 16 + rb + rr;
                size_t idx = (size_t)row * N + col;
                float v = acc[i][j][rr] + bf2f(Self[idx]);
                v = (v >= 0.f) ? v : 0.1f * v;   // leaky ReLU
                if (Yb) Yb[idx] = f2bf(v);
                if (Yf) Yf[idx] = v;
            }
        }
}

// ---------------------------------------------------------------------------
// Row L2-normalize: out = x / (||x||_2 + 1e-9).  One wave32 per row, N=256.
// ---------------------------------------------------------------------------
__global__ __launch_bounds__(256) void norm_kernel(
    const float* __restrict__ X, float* __restrict__ Out, int N)
{
    int row  = blockIdx.x * 8 + (threadIdx.x >> 5);
    int lane = threadIdx.x & 31;
    const float* x = X + (size_t)row * N;
    float s = 0.f;
    for (int c = lane; c < N; c += 32) { float v = x[c]; s += v * v; }
#pragma unroll
    for (int m = 16; m > 0; m >>= 1) s += __shfl_xor(s, m, 32);
    float inv = 1.0f / (sqrtf(s) + 1e-9f);
    float* o = Out + (size_t)row * N;
    for (int c = lane; c < N; c += 32) o[c] = x[c] * inv;
}

// ---------------------------------------------------------------------------
extern "C" void kernel_launch(void* const* d_in, const int* in_sizes, int n_in,
                              void* d_out, int out_size, void* d_ws, size_t ws_size,
                              hipStream_t stream)
{
    (void)in_sizes; (void)n_in; (void)out_size; (void)ws_size;
    constexpr int NP = 8192, NQ = 8192, DP = 512, D1 = 256, D2 = 256;

    const float* hp         = (const float*)d_in[0];
    const float* hq         = (const float*)d_in[1];
    const float* a_cons     = (const float*)d_in[2];
    const float* a_prod     = (const float*)d_in[3];
    const float* a_rev_cons = (const float*)d_in[4];
    const float* a_rev_prod = (const float*)d_in[5];
    // order: wp1 wq1 wcons1 wrprod1 wprod1 wrcons1 wp2 wq2 wcons2 wrprod2 wprod2 wrcons2
    const float* Wf[12]; const float* Bf[12];
    for (int i = 0; i < 12; i++) { Wf[i] = (const float*)d_in[6 + 2*i];
                                   Bf[i] = (const float*)d_in[7 + 2*i]; }

    char* base = (char*)d_ws; size_t off = 0;
    auto carve = [&](size_t bytes) -> void* {
        void* p = base + off; off = (off + bytes + 255) & ~(size_t)255; return p;
    };
    u16t* hpb = (u16t*)carve((size_t)NP * DP * 2);   // 8 MB
    u16t* hqb = (u16t*)carve((size_t)NQ * DP * 2);   // 8 MB
    u16t* Wb[12];
    for (int i = 0; i < 12; i++)
        Wb[i] = (u16t*)carve((size_t)(i < 6 ? D1 * DP : D2 * D1) * 2);
    u16t* lout[6];
    for (int i = 0; i < 6; i++) lout[i] = (u16t*)carve((size_t)NP * D1 * 2);
    u16t* hp1b = (u16t*)carve((size_t)NP * D1 * 2);
    u16t* hq1b = (u16t*)carve((size_t)NQ * D1 * 2);
    // f32 pre-norm outputs alias the (dead by then) bf16 input copies: 8 MB each
    float* hp2f = (float*)hpb;
    float* hq2f = (float*)hqb;

    // ---- casts to bf16 ----
    cast_bf16_kernel<<<2048, 256, 0, stream>>>(hp, hpb, (size_t)NP * DP);
    cast_bf16_kernel<<<2048, 256, 0, stream>>>(hq, hqb, (size_t)NQ * DP);
    for (int i = 0; i < 12; i++) {
        size_t n = (size_t)(i < 6 ? D1 * DP : D2 * D1);
        cast_bf16_kernel<<<256, 256, 0, stream>>>(Wf[i], Wb[i], n);
    }

    dim3 g(NP / BM), b(NT);   // 128 blocks x 4 waves

    // ---- layer 1 linears (K = 512) ----
    // self terms row-major; aggregation operands N-major (colmajor=1)
    lin_kernel<<<g, b, 0, stream>>>(hpb, Wb[0], Bf[0], lout[0], NP, D1, DP, 0); // selfP
    lin_kernel<<<g, b, 0, stream>>>(hqb, Wb[1], Bf[1], lout[1], NQ, D1, DP, 0); // selfQ
    lin_kernel<<<g, b, 0, stream>>>(hqb, Wb[2], Bf[2], lout[2], NQ, D1, DP, 1); // cons(hq)
    lin_kernel<<<g, b, 0, stream>>>(hqb, Wb[3], Bf[3], lout[3], NQ, D1, DP, 1); // rprod(hq)
    lin_kernel<<<g, b, 0, stream>>>(hpb, Wb[4], Bf[4], lout[4], NP, D1, DP, 1); // prod(hp)
    lin_kernel<<<g, b, 0, stream>>>(hpb, Wb[5], Bf[5], lout[5], NP, D1, DP, 1); // rcons(hp)

    // ---- layer 1 aggregation (K = 8192) ----
    agg_kernel<<<g, b, 0, stream>>>(a_cons, a_rev_prod, lout[2], lout[3], lout[0],
                                    hp1b, (float*)nullptr, NP, D1, NQ);
    agg_kernel<<<g, b, 0, stream>>>(a_prod, a_rev_cons, lout[4], lout[5], lout[1],
                                    hq1b, (float*)nullptr, NQ, D1, NP);

    // ---- layer 2 linears (K = 256), reuse lout buffers ----
    lin_kernel<<<g, b, 0, stream>>>(hp1b, Wb[6],  Bf[6],  lout[0], NP, D2, D1, 0); // selfP2
    lin_kernel<<<g, b, 0, stream>>>(hq1b, Wb[7],  Bf[7],  lout[1], NQ, D2, D1, 0); // selfQ2
    lin_kernel<<<g, b, 0, stream>>>(hq1b, Wb[8],  Bf[8],  lout[2], NQ, D2, D1, 1); // cons2
    lin_kernel<<<g, b, 0, stream>>>(hq1b, Wb[9],  Bf[9],  lout[3], NQ, D2, D1, 1); // rprod2
    lin_kernel<<<g, b, 0, stream>>>(hp1b, Wb[10], Bf[10], lout[4], NP, D2, D1, 1); // prod2
    lin_kernel<<<g, b, 0, stream>>>(hp1b, Wb[11], Bf[11], lout[5], NP, D2, D1, 1); // rcons2

    // ---- layer 2 aggregation, f32 output for normalization ----
    agg_kernel<<<g, b, 0, stream>>>(a_cons, a_rev_prod, lout[2], lout[3], lout[0],
                                    (u16t*)nullptr, hp2f, NP, D2, NQ);
    agg_kernel<<<g, b, 0, stream>>>(a_prod, a_rev_cons, lout[4], lout[5], lout[1],
                                    (u16t*)nullptr, hq2f, NQ, D2, NP);

    // ---- row normalize into d_out (hp2 first, then hq2) ----
    norm_kernel<<<NP / 8, 256, 0, stream>>>(hp2f, (float*)d_out, D2);
    norm_kernel<<<NQ / 8, 256, 0, stream>>>(hq2f, (float*)d_out + (size_t)NP * D2, D2);
}
